// GraphConvNet_Regression_34668976013313
// MI455X (gfx1250) — matrix operand order repaired and checked by reading, not compiled, not run on hardware
//
#include <hip/hip_runtime.h>
#include <hip/hip_bf16.h>
#include <stdint.h>

#define WGS   256
#define NN    8192
#define DD    2000
#define DP    2016
#define HH0   256
#define HH1   256
#define HH2   128
#define NBINS 8192
#define QIDX  66437774LL   // round(0.99 * (8192*8192 - 1))
#define LP    20           // LDS row pitch in dwords (16B-aligned rows, conflict-free gathers)

typedef __attribute__((ext_vector_type(16))) __bf16 v16bf;
typedef __attribute__((ext_vector_type(8)))  float  v8f;

__device__ __forceinline__ unsigned short f2bf(float f) {
  unsigned int u = __float_as_uint(f);
  u += 0x7fffu + ((u >> 16) & 1u);           // round-to-nearest-even
  return (unsigned short)(u >> 16);
}
__device__ __forceinline__ float bf2f(unsigned short h) {
  return __uint_as_float(((unsigned int)h) << 16);
}

// ---------------- workspace layout ----------------
static constexpr size_t alup(size_t x) { return (x + 255) & ~(size_t)255; }
static constexpr size_t OFF_XN   = 0;                                    // xn bf16 [NN,DP]
static constexpr size_t OFF_XB   = OFF_XN   + alup((size_t)NN*DP*2);     // x  bf16 [NN,DP]
static constexpr size_t OFF_WIN  = OFF_XB   + alup((size_t)NN*DP*2);     // w_in bf16 [256,DP]
static constexpr size_t OFF_W1L  = OFF_WIN  + alup((size_t)HH0*DP*2);
static constexpr size_t OFF_W1R  = OFF_W1L  + alup((size_t)HH1*HH0*2);
static constexpr size_t OFF_W2L  = OFF_W1R  + alup((size_t)HH1*HH0*2);
static constexpr size_t OFF_W2R  = OFF_W2L  + alup((size_t)HH2*HH1*2);
static constexpr size_t OFF_H0B  = OFF_W2R  + alup((size_t)HH2*HH1*2);   // h0 bf16 [NN,256]
static constexpr size_t OFF_MEAN = OFF_H0B  + alup((size_t)NN*HH0*2);    // mean bf16 [NN,256]
static constexpr size_t OFF_H1B  = OFF_MEAN + alup((size_t)NN*256*2);    // h1 bf16 [NN,256]
static constexpr size_t OFF_TMP  = OFF_H1B  + alup((size_t)NN*HH1*2);    // tmp f32 [NN,256]
static constexpr size_t OFF_TMP2 = OFF_TMP  + alup((size_t)NN*HH1*4);    // tmp2 f32 [NN,128]
static constexpr size_t OFF_H2F  = OFF_TMP2 + alup((size_t)NN*HH2*4);    // h2 f32 [NN,128]
static constexpr size_t OFF_ADJ  = OFF_H2F  + alup((size_t)NN*HH2*4);    // adj bits [NN, NN/32] u32
static constexpr size_t OFF_DEG  = OFF_ADJ  + alup((size_t)NN*(NN/32)*4);
static constexpr size_t OFF_HST1 = OFF_DEG  + alup((size_t)NN*4);
static constexpr size_t OFF_HST2 = OFF_HST1 + alup((size_t)NBINS*4);
static constexpr size_t OFF_SCAL = OFF_HST2 + alup((size_t)NBINS*4);
static constexpr size_t WS_NEED  = OFF_SCAL + 1024;

// ---------------- small kernels ----------------
__global__ void zero_u32_kernel(unsigned int* p, long long n) {
  long long i = (long long)blockIdx.x * blockDim.x + threadIdx.x;
  long long stride = (long long)gridDim.x * blockDim.x;
  for (; i < n; i += stride) p[i] = 0u;
}

__global__ void init_scalars_kernel(float* fs, int* is) {
  fs[0] = 0.0f;            // range1.lo
  fs[1] = (float)NBINS;    // range1.scale
  is[0] = 0;               // below = 0 for stage 1
}

__global__ __launch_bounds__(WGS)
void rownorm_kernel(const float* __restrict__ x,
                    unsigned short* __restrict__ xb,
                    unsigned short* __restrict__ xnb) {
  __shared__ float red[WGS];
  int i = blockIdx.x, tid = threadIdx.x;
  const float* xr = x + (size_t)i * DD;
  float s = 0.f;
  for (int k = tid; k < DD; k += WGS) { float v = xr[k]; s += v * v; }
  red[tid] = s; __syncthreads();
  for (int off = WGS / 2; off > 0; off >>= 1) {
    if (tid < off) red[tid] += red[tid + off];
    __syncthreads();
  }
  float inv = 1.0f / fmaxf(sqrtf(red[0]), 1e-8f);
  for (int k = tid; k < DP; k += WGS) {
    float v = (k < DD) ? xr[k] : 0.f;
    xb [(size_t)i * DP + k] = f2bf(v);
    xnb[(size_t)i * DP + k] = f2bf(v * inv);
  }
}

__global__ void cvt_w_kernel(const float* __restrict__ src,
                             unsigned short* __restrict__ dst, int C, int Cp) {
  int r = blockIdx.x, tid = threadIdx.x;
  for (int k = tid; k < Cp; k += blockDim.x)
    dst[(size_t)r * Cp + k] = f2bf(k < C ? src[(size_t)r * C + k] : 0.f);
}

// Find first bin b with cum >= (qidx - *belowIn + 1); emit refined range / eps.
__global__ __launch_bounds__(WGS)
void scan_kernel(const unsigned int* __restrict__ hist,
                 const float* __restrict__ rangeIn, const int* __restrict__ belowIn,
                 float* rangeOut, int* belowOut, float* epsOut, long long qidx) {
  __shared__ long long part[WGS];
  int tid = threadIdx.x;
  long long local = 0;
  for (int i = 0; i < NBINS / WGS; i++) local += hist[tid * (NBINS / WGS) + i];
  part[tid] = local; __syncthreads();
  if (tid == 0) {
    long long run = 0;
    for (int i = 0; i < WGS; i++) { long long t = part[i]; part[i] = run; run += t; }
  }
  __syncthreads();
  long long run = part[tid];
  long long below = (long long)(*belowIn);
  long long target = qidx - below + 1;
  float lo = rangeIn[0], scale = rangeIn[1];
  for (int i = 0; i < NBINS / WGS; i++) {
    long long cnt = hist[tid * (NBINS / WGS) + i];
    if (run < target && run + cnt >= target) {
      int b = tid * (NBINS / WGS) + i;
      float binw = 1.0f / scale;
      float newlo = lo + (float)b * binw;
      rangeOut[0] = newlo;
      rangeOut[1] = scale * (float)NBINS;
      *belowOut = (int)(below + run);
      *epsOut = newlo;
    }
    run += cnt;
  }
}

__global__ void deg_kernel(const unsigned int* __restrict__ adj, float* __restrict__ deg) {
  int i = blockIdx.x * blockDim.x + threadIdx.x;
  if (i < NN) {
    const unsigned int* r = adj + (size_t)i * (NN / 32);
    int s = 0;
    for (int w = 0; w < NN / 32; w++) s += __popc(r[w]);
    deg[i] = fmaxf((float)s, 1.0f);
  }
}

// mean_i = (1/deg_i) * sum_{j: adj[i][j]} h[j]   (adj symmetric), 256 channels
__global__ __launch_bounds__(WGS)
void spmm_mean_kernel(const unsigned int* __restrict__ adj,
                      const unsigned short* __restrict__ h,
                      const float* __restrict__ deg,
                      unsigned short* __restrict__ meanb) {
  __shared__ unsigned int words[NN / 32];
  int i = blockIdx.x, tid = threadIdx.x;
  for (int w = tid; w < NN / 32; w += WGS) words[w] = adj[(size_t)i * (NN / 32) + w];
  __syncthreads();
  float acc = 0.f;
  int c = tid;  // WGS == 256 channels
  for (int w = 0; w < NN / 32; w++) {
    unsigned int m = words[w];
    while (m) {
      int b = __ffs((int)m) - 1;
      m &= m - 1;
      int j = (w << 5) + b;
      acc += bf2f(h[(size_t)j * 256 + c]);
    }
  }
  meanb[(size_t)i * 256 + c] = f2bf(acc / deg[i]);
}

__global__ void out_kernel(const float* __restrict__ h2, const float* __restrict__ wout,
                           const float* __restrict__ bout, float* __restrict__ out) {
  int i = blockIdx.x * blockDim.x + threadIdx.x;
  if (i < NN) {
    float a = bout[0];
    const float* hr = h2 + (size_t)i * HH2;
    for (int k = 0; k < HH2; k++) a += hr[k] * wout[k];
    out[i] = a;
  }
}

// ---------------- WMMA bf16 GEMM: C = A[M,Kp] * B[N,Kp]^T ----------------
// WG = 8 waves (4 in M x 2 in N); wave computes 32x64 via 2x4 wmma tiles.
// WG tile 128(M) x 128(N); M%128==0, N%128==0, Kp%32==0 required.
enum { MODE_STORE = 0, MODE_HIST = 1, MODE_ADJ = 2 };

template <int MODE>
__global__ __launch_bounds__(WGS)
void gemm_bf16_kernel(const unsigned short* __restrict__ A,
                      const unsigned short* __restrict__ B,
                      int N, int Kp,
                      float* __restrict__ outF, unsigned short* __restrict__ outH,
                      const float* __restrict__ bias, const float* __restrict__ addend,
                      int relu,
                      const float* __restrict__ range, unsigned int* __restrict__ ghist,
                      const float* __restrict__ epsPtr, unsigned int* __restrict__ adjbits) {
  __shared__ int As[128 * LP];   // 128 rows x 16 dwords (32 bf16), pitch LP=20 (16B-aligned rows)
  __shared__ int Bs[128 * LP];
  __shared__ unsigned int lh[(MODE == MODE_HIST) ? NBINS : 1];
  int tid = threadIdx.x;
  if (MODE == MODE_HIST)
    for (int b = tid; b < NBINS; b += WGS) lh[b] = 0u;

  const int tileM = blockIdx.y, tileN = blockIdx.x;
  const int w = tid >> 5, lane = tid & 31;
  const int waveM = w >> 1, waveN = w & 1;          // 4 x 2 wave grid
  const int rowBase = tileM * 128, colBase = tileN * 128;
  const int lda = Kp >> 1;  // dwords per row
  const int* Ag = (const int*)A;
  const int* Bg = (const int*)B;

  union Frag { int u[8]; int4 q[2]; v16bf v; };
  Frag afrag[2], bfrag[4];
  union { float f[8]; v8f v; } c[2][4];
#pragma unroll
  for (int tm = 0; tm < 2; tm++)
#pragma unroll
    for (int tn = 0; tn < 4; tn++)
#pragma unroll
      for (int k = 0; k < 8; k++) c[tm][tn].f[k] = 0.f;

  const int lr = lane & 15;
  const int aq = (lane < 16) ? 0 : 1;   // A frag: int4 pieces {aq, 2+aq} (ISA K-half interleave)
  const int bq = (lane < 16) ? 0 : 1;   // B frag: int4 pieces {bq, 2+bq} (ISA K-half interleave)
  const int ldr = tid >> 2;             // loader row (0..63)
  const int ldq = (tid & 3) * 4;        // loader dword quad

  for (int k0 = 0; k0 < Kp; k0 += 32) {
    __syncthreads();
#pragma unroll
    for (int hh = 0; hh < 2; hh++) {    // A panel: 128 rows x 4 int4
      int r = ldr + hh * 64;
      const int4 d = *(const int4*)(Ag + (size_t)(rowBase + r) * lda + (k0 >> 1) + ldq);
      *(int4*)&As[r * LP + ldq] = d;
    }
#pragma unroll
    for (int hh = 0; hh < 2; hh++) {    // B panel: 128 rows x 4 int4
      int r = ldr + hh * 64;
      const int4 d = *(const int4*)(Bg + (size_t)(colBase + r) * lda + (k0 >> 1) + ldq);
      *(int4*)&Bs[r * LP + ldq] = d;
    }
    __syncthreads();
#pragma unroll
    for (int tm = 0; tm < 2; tm++) {
      const int4* p = (const int4*)&As[(waveM * 32 + tm * 16 + lr) * LP];
      afrag[tm].q[0] = p[aq];
      afrag[tm].q[1] = p[2 + aq];
    }
#pragma unroll
    for (int tn = 0; tn < 4; tn++) {
      const int4* p = (const int4*)&Bs[(waveN * 64 + tn * 16 + lr) * LP];
      bfrag[tn].q[0] = p[bq];
      bfrag[tn].q[1] = p[2 + bq];
    }
#pragma unroll
    for (int tm = 0; tm < 2; tm++)
#pragma unroll
      for (int tn = 0; tn < 4; tn++)
        c[tm][tn].v = __builtin_amdgcn_wmma_f32_16x16x32_bf16(
            false, afrag[tm].v, false, bfrag[tn].v, (short)0, c[tm][tn].v, false, false);
  }

  // epilogue: C layout — VGPR v: M = v + 8*(lane>=16), N = lane%16
  float lo = 0.f, scale = 0.f, eps = 0.f;
  if (MODE == MODE_HIST) { lo = range[0]; scale = range[1]; }
  if (MODE == MODE_ADJ)  { eps = *epsPtr; }
  const int rOff = (lane >= 16) ? 8 : 0;
#pragma unroll
  for (int tm = 0; tm < 2; tm++)
#pragma unroll
    for (int tn = 0; tn < 4; tn++)
#pragma unroll
      for (int v = 0; v < 8; v++) {
        int row = rowBase + waveM * 32 + tm * 16 + v + rOff;
        int col = colBase + waveN * 64 + tn * 16 + lr;
        float val = c[tm][tn].f[v];
        if (MODE == MODE_STORE) {
          if (bias)   val += bias[col];
          if (addend) val += addend[(size_t)row * N + col];
          if (relu)   val = fmaxf(val, 0.f);
          if (outF) outF[(size_t)row * N + col] = val;
          if (outH) outH[(size_t)row * N + col] = f2bf(val);
        } else if (MODE == MODE_HIST) {
          float vv = fabsf(val);
          if (row == col) vv = 0.f;             // diagonal zeroed, still counted
          int b = (int)floorf((vv - lo) * scale);
          if (b >= 0 && b < NBINS) atomicAdd(&lh[b], 1u);
        } else {  // MODE_ADJ
          float vv = fabsf(val);
          if (row != col && vv >= eps)
            atomicOr(&adjbits[(size_t)row * (NN / 32) + (col >> 5)], 1u << (col & 31));
        }
      }
  if (MODE == MODE_HIST) {
    __syncthreads();
    for (int b = tid; b < NBINS; b += WGS)
      if (lh[b]) atomicAdd(&ghist[b], lh[b]);
  }
}

// ---------------- launcher ----------------
extern "C" void kernel_launch(void* const* d_in, const int* in_sizes, int n_in,
                              void* d_out, int out_size, void* d_ws, size_t ws_size,
                              hipStream_t stream) {
  (void)in_sizes; (void)n_in; (void)out_size;
  if (ws_size < WS_NEED) return;

  const float* x    = (const float*)d_in[0];
  const float* w_in = (const float*)d_in[1];
  const float* w1l  = (const float*)d_in[2];
  const float* b1l  = (const float*)d_in[3];
  const float* w1r  = (const float*)d_in[4];
  const float* w2l  = (const float*)d_in[5];
  const float* b2l  = (const float*)d_in[6];
  const float* w2r  = (const float*)d_in[7];
  const float* wout = (const float*)d_in[8];
  const float* bout = (const float*)d_in[9];
  float* out = (float*)d_out;
  char* ws = (char*)d_ws;

  unsigned short* xnb   = (unsigned short*)(ws + OFF_XN);
  unsigned short* xb    = (unsigned short*)(ws + OFF_XB);
  unsigned short* winb  = (unsigned short*)(ws + OFF_WIN);
  unsigned short* w1lb  = (unsigned short*)(ws + OFF_W1L);
  unsigned short* w1rb  = (unsigned short*)(ws + OFF_W1R);
  unsigned short* w2lb  = (unsigned short*)(ws + OFF_W2L);
  unsigned short* w2rb  = (unsigned short*)(ws + OFF_W2R);
  unsigned short* h0b   = (unsigned short*)(ws + OFF_H0B);
  unsigned short* meanb = (unsigned short*)(ws + OFF_MEAN);
  unsigned short* h1b   = (unsigned short*)(ws + OFF_H1B);
  float* tmpf  = (float*)(ws + OFF_TMP);
  float* tmp2f = (float*)(ws + OFF_TMP2);
  float* h2f   = (float*)(ws + OFF_H2F);
  unsigned int* adjbits = (unsigned int*)(ws + OFF_ADJ);
  float* deg   = (float*)(ws + OFF_DEG);
  unsigned int* hist1 = (unsigned int*)(ws + OFF_HST1);
  unsigned int* hist2 = (unsigned int*)(ws + OFF_HST2);
  float* fs = (float*)(ws + OFF_SCAL);       // fs[0..1]=range1, fs[2..3]=range2, fs[4..5]=scratch, fs[6]=eps
  int*   is = (int*)(ws + OFF_SCAL + 512);   // is[0]=0, is[1]=below1, is[2]=scratch

  // prep: normalize + bf16 conversions, zero accumulators
  rownorm_kernel<<<NN, WGS, 0, stream>>>(x, xb, xnb);
  cvt_w_kernel<<<HH0, WGS, 0, stream>>>(w_in, winb, DD, DP);
  cvt_w_kernel<<<HH1, WGS, 0, stream>>>(w1l, w1lb, HH0, HH0);
  cvt_w_kernel<<<HH1, WGS, 0, stream>>>(w1r, w1rb, HH0, HH0);
  cvt_w_kernel<<<HH2, WGS, 0, stream>>>(w2l, w2lb, HH1, HH1);
  cvt_w_kernel<<<HH2, WGS, 0, stream>>>(w2r, w2rb, HH1, HH1);
  zero_u32_kernel<<<2048, WGS, 0, stream>>>(adjbits, (long long)NN * (NN / 32));
  zero_u32_kernel<<<32, WGS, 0, stream>>>(hist1, NBINS);
  zero_u32_kernel<<<32, WGS, 0, stream>>>(hist2, NBINS);
  init_scalars_kernel<<<1, 1, 0, stream>>>(fs, is);

  // quantile threshold via two histogram passes over recomputed sim tiles
  dim3 gSim(NN / 128, NN / 128);
  gemm_bf16_kernel<MODE_HIST><<<gSim, WGS, 0, stream>>>(xnb, xnb, NN, DP,
      nullptr, nullptr, nullptr, nullptr, 0, fs + 0, hist1, nullptr, nullptr);
  scan_kernel<<<1, WGS, 0, stream>>>(hist1, fs + 0, is + 0, fs + 2, is + 1, fs + 6, QIDX);
  gemm_bf16_kernel<MODE_HIST><<<gSim, WGS, 0, stream>>>(xnb, xnb, NN, DP,
      nullptr, nullptr, nullptr, nullptr, 0, fs + 2, hist2, nullptr, nullptr);
  scan_kernel<<<1, WGS, 0, stream>>>(hist2, fs + 2, is + 1, fs + 4, is + 2, fs + 6, QIDX);

  // adjacency bitmask + degrees
  gemm_bf16_kernel<MODE_ADJ><<<gSim, WGS, 0, stream>>>(xnb, xnb, NN, DP,
      nullptr, nullptr, nullptr, nullptr, 0, nullptr, nullptr, fs + 6, adjbits);
  deg_kernel<<<NN / WGS, WGS, 0, stream>>>(adjbits, deg);

  // h0 = relu(x @ w_in^T)
  dim3 g256(HH0 / 128, NN / 128);
  gemm_bf16_kernel<MODE_STORE><<<g256, WGS, 0, stream>>>(xb, winb, HH0, DP,
      nullptr, h0b, nullptr, nullptr, 1, nullptr, nullptr, nullptr, nullptr);

  // SAGE layer 1: h1 = relu(mean1 @ w1l^T + b1 + h0 @ w1r^T)
  gemm_bf16_kernel<MODE_STORE><<<g256, WGS, 0, stream>>>(h0b, w1rb, HH1, HH0,
      tmpf, nullptr, nullptr, nullptr, 0, nullptr, nullptr, nullptr, nullptr);
  spmm_mean_kernel<<<NN, WGS, 0, stream>>>(adjbits, h0b, deg, meanb);
  gemm_bf16_kernel<MODE_STORE><<<g256, WGS, 0, stream>>>(meanb, w1lb, HH1, HH0,
      nullptr, h1b, b1l, tmpf, 1, nullptr, nullptr, nullptr, nullptr);

  // SAGE layer 2: h2 = relu(mean2 @ w2l^T + b2 + h1 @ w2r^T)
  dim3 g128(HH2 / 128, NN / 128);
  gemm_bf16_kernel<MODE_STORE><<<g128, WGS, 0, stream>>>(h1b, w2rb, HH2, HH1,
      tmp2f, nullptr, nullptr, nullptr, 0, nullptr, nullptr, nullptr, nullptr);
  spmm_mean_kernel<<<NN, WGS, 0, stream>>>(adjbits, h1b, deg, meanb);
  gemm_bf16_kernel<MODE_STORE><<<g128, WGS, 0, stream>>>(meanb, w2lb, HH2, HH1,
      h2f, nullptr, b2l, tmp2f, 1, nullptr, nullptr, nullptr, nullptr);

  // out = h2 @ w_out^T + b_out
  out_kernel<<<NN / WGS, WGS, 0, stream>>>(h2f, wout, bout, out);
}
